// VisualLlamaBlock_56813827392249
// MI455X (gfx1250) — matrix operand order repaired and checked by reading, not compiled, not run on hardware
//
#include <hip/hip_runtime.h>
#include <hip/hip_bf16.h>

// ---------------------------------------------------------------------------
// Types for CDNA5 WMMA (wave32, 16x16x32 bf16 -> f32)
// ---------------------------------------------------------------------------
typedef __attribute__((ext_vector_type(16))) __bf16 v16bf;
typedef __attribute__((ext_vector_type(4)))  __bf16 v4bf;
typedef __attribute__((ext_vector_type(8)))  float  v8f;
typedef __attribute__((ext_vector_type(4)))  float  f32x4;
typedef __attribute__((ext_vector_type(4)))  unsigned int u32x4;

union Frag16 { v16bf v; u32x4 u[2]; };

// ---------------------------------------------------------------------------
// CDNA5 async global->LDS copy (ASYNCcnt path), with sync fallback.
// Builtin signature (from hipcc diagnostic): param0 = int4 AS1* (global src),
// param1 = int4 AS3* (LDS dst), then imm offset, imm cpol.
// ---------------------------------------------------------------------------
#if defined(__HIP_DEVICE_COMPILE__) && \
    __has_builtin(__builtin_amdgcn_global_load_async_to_lds_b128) && \
    __has_builtin(__builtin_amdgcn_s_wait_asynccnt)
#define HAVE_ASYNC_LDS 1
#else
#define HAVE_ASYNC_LDS 0
#endif

typedef int v4i __attribute__((vector_size(16)));
typedef __attribute__((address_space(1))) v4i* glb_i4p;
typedef __attribute__((address_space(3))) v4i* lds_i4p;

__device__ inline void async_cp16(void* dst_lds, const void* src_glob) {
#if HAVE_ASYNC_LDS
    __builtin_amdgcn_global_load_async_to_lds_b128(
        (glb_i4p)src_glob, (lds_i4p)dst_lds, 0, 0);
#else
    *(u32x4*)dst_lds = *(const u32x4*)src_glob;
#endif
}
__device__ inline void async_wait_all() {
#if HAVE_ASYNC_LDS
    __builtin_amdgcn_s_wait_asynccnt(0);
#endif
}

// A-matrix fragment (16x32 bf16): lane<16 holds M=lane, K chunks {0..7},{16..23};
// lane>=16 holds M=lane-16, K chunks {8..15},{24..31}.  `base` = row ptr at K=0,
// row is k-contiguous.
__device__ inline v16bf load_frag_a(const __bf16* base, int half) {
    Frag16 f;
    f.u[0] = *(const u32x4*)(base + 8 * half);
    f.u[1] = *(const u32x4*)(base + 16 + 8 * half);
    return f.v;
}
// B-matrix fragment (32x16 bf16): lane holds column n=lane&15; lanes<16 carry
// K chunks {0..7},{16..23}, lanes>=16 carry {8..15},{24..31} (k-contiguous row layout [n][k]).
__device__ inline v16bf load_frag_b(const __bf16* base, int half) {
    Frag16 f;
    f.u[0] = *(const u32x4*)(base + 8 * half);
    f.u[1] = *(const u32x4*)(base + 16 + 8 * half);
    return f.v;
}

#define WMMA_BF16(A, B, C) \
    __builtin_amdgcn_wmma_f32_16x16x32_bf16(false, (A), false, (B), (short)0, (C), false, false)

// fast tanh via exp (identical identity, no branchy libm path)
__device__ inline float fast_tanh(float x) {
    float e = __expf(2.f * x);
    return 1.f - 2.f / (e + 1.f);
}

// ---------------------------------------------------------------------------
// Generic tiled GEMM:  out[M,N] = A(bf16)[M,K] * B(W fp32, cvt->bf16) (+epilogue)
// BT=true : W is [N,K] row-major (nn.Linear, y = x @ W^T)
// BT=false: W is [K,N] row-major (Conv1D,    y = x @ W)
// ---------------------------------------------------------------------------
template <bool BT, bool BIAS, bool ACC, bool RES, bool STF, bool STB>
__global__ __launch_bounds__(256) void gemm_k(
    const __bf16* __restrict__ A, int lda,
    const float*  __restrict__ Bw, int ldb,
    const float*  __restrict__ bias,
    const float*  __restrict__ resid,
    const float*  __restrict__ accIn,
    float* __restrict__ outF, __bf16* __restrict__ outB,
    int M, int N, int K)
{
    constexpr int BM = 128, BN = 128, BK = 32;
    __shared__ __bf16 As[BM * BK];   // [m][k]
    __shared__ __bf16 Bs[BN * BK];   // [n][k]
    const int tid = threadIdx.x;
    const int wid = tid >> 5, lane = tid & 31;
    const int half = lane >> 4, hl = lane & 15;
    const int m0 = blockIdx.y * BM, n0 = blockIdx.x * BN;
    const int wm = wid >> 2, wn = wid & 3;   // 2 x 4 wave grid, 64x32 per wave

    v8f acc[4][2] = {};

    for (int k0 = 0; k0 < K; k0 += BK) {
        // --- stage A tile (bf16): async global->LDS, 512 x 16B chunks ---
#pragma unroll
        for (int it = 0; it < 2; ++it) {
            int id = it * 256 + tid;
            int r = id >> 2, c = id & 3;
            async_cp16(&As[r * BK + c * 8],
                       A + (size_t)(m0 + r) * lda + k0 + c * 8);
        }
        // --- stage B tile (fp32 -> bf16 conversion through VGPRs) ---
        if (BT) {
#pragma unroll
            for (int it = 0; it < 4; ++it) {
                int id = it * 256 + tid;
                int r = id >> 3, c = id & 7;
                const float* src = Bw + (size_t)(n0 + r) * ldb + k0 + c * 4;
                f32x4 f = *(const f32x4*)src;
                if (k0 + BK < K) __builtin_prefetch(src + BK, 0, 3);
                v4bf p;
                p[0] = (__bf16)f[0]; p[1] = (__bf16)f[1];
                p[2] = (__bf16)f[2]; p[3] = (__bf16)f[3];
                *(v4bf*)&Bs[r * BK + c * 4] = p;
            }
        } else {
#pragma unroll
            for (int it = 0; it < 4; ++it) {
                int id = it * 256 + tid;
                int kk = id >> 5, c = id & 31;
                const float* src = Bw + (size_t)(k0 + kk) * ldb + n0 + c * 4;
                f32x4 f = *(const f32x4*)src;
                if (k0 + BK < K) __builtin_prefetch(src + (size_t)BK * ldb, 0, 3);
#pragma unroll
                for (int j = 0; j < 4; ++j)
                    Bs[(c * 4 + j) * BK + kk] = (__bf16)f[j];
            }
        }
        async_wait_all();
        __syncthreads();

        // --- compute ---
        v16bf af[4], bf[2];
#pragma unroll
        for (int i = 0; i < 4; ++i)
            af[i] = load_frag_a(&As[(wm * 64 + i * 16 + hl) * BK], half);
#pragma unroll
        for (int j = 0; j < 2; ++j)
            bf[j] = load_frag_b(&Bs[(wn * 32 + j * 16 + hl) * BK], half);
#pragma unroll
        for (int i = 0; i < 4; ++i)
#pragma unroll
            for (int j = 0; j < 2; ++j)
                acc[i][j] = WMMA_BF16(af[i], bf[j], acc[i][j]);
        __syncthreads();
    }

    // --- epilogue: C layout lane(n)=lane&15, row m = 8*(lane>>4)+r ---
#pragma unroll
    for (int i = 0; i < 4; ++i) {
#pragma unroll
        for (int j = 0; j < 2; ++j) {
            int n = n0 + wn * 32 + j * 16 + hl;
            float bv = BIAS ? bias[n] : 0.f;
#pragma unroll
            for (int r = 0; r < 8; ++r) {
                int m = m0 + wm * 64 + i * 16 + half * 8 + r;
                size_t off = (size_t)m * N + n;
                float v = acc[i][j][r] + bv;
                if (ACC) v += accIn[off];
                if (RES) v += resid[off];
                if (STF) outF[off] = v;
                if (STB) outB[off] = (__bf16)v;
            }
        }
    }
}

// ---------------------------------------------------------------------------
// Dual GEMM for MLP: prod = gelu(h2 @ Wg^T) * (h2 @ Wu^T), both W [N,K]
// ---------------------------------------------------------------------------
__global__ __launch_bounds__(256) void dualgemm_k(
    const __bf16* __restrict__ A, int lda,
    const float* __restrict__ Wg, const float* __restrict__ Wu, int ldb,
    __bf16* __restrict__ outP, int M, int N, int K)
{
    constexpr int BM = 128, BN = 64, BK = 32;
    __shared__ __bf16 As[BM * BK];
    __shared__ __bf16 Bg[BN * BK];
    __shared__ __bf16 Bu[BN * BK];
    const int tid = threadIdx.x;
    const int wid = tid >> 5, lane = tid & 31;
    const int half = lane >> 4, hl = lane & 15;
    const int m0 = blockIdx.y * BM, n0 = blockIdx.x * BN;
    const int wm = wid >> 1, wn = wid & 1;   // 4 x 2 wave grid, 32x32 per wave

    v8f ag[2][2] = {}, au[2][2] = {};

    for (int k0 = 0; k0 < K; k0 += BK) {
#pragma unroll
        for (int it = 0; it < 2; ++it) {
            int id = it * 256 + tid;
            int r = id >> 2, c = id & 3;
            async_cp16(&As[r * BK + c * 8],
                       A + (size_t)(m0 + r) * lda + k0 + c * 8);
        }
#pragma unroll
        for (int it = 0; it < 4; ++it) {
            int id = it * 256 + tid;
            const float* Wsrc = (id < 512) ? Wg : Wu;
            __bf16*      Bdst = (id < 512) ? Bg : Bu;
            int id2 = id & 511;
            int r = id2 >> 3, c = id2 & 7;
            const float* src = Wsrc + (size_t)(n0 + r) * ldb + k0 + c * 4;
            f32x4 f = *(const f32x4*)src;
            if (k0 + BK < K) __builtin_prefetch(src + BK, 0, 3);
            v4bf p;
            p[0] = (__bf16)f[0]; p[1] = (__bf16)f[1];
            p[2] = (__bf16)f[2]; p[3] = (__bf16)f[3];
            *(v4bf*)&Bdst[r * BK + c * 4] = p;
        }
        async_wait_all();
        __syncthreads();

        v16bf af[2];
#pragma unroll
        for (int i = 0; i < 2; ++i)
            af[i] = load_frag_a(&As[(wm * 32 + i * 16 + hl) * BK], half);
#pragma unroll
        for (int j = 0; j < 2; ++j) {
            v16bf bg = load_frag_b(&Bg[(wn * 32 + j * 16 + hl) * BK], half);
            v16bf bu = load_frag_b(&Bu[(wn * 32 + j * 16 + hl) * BK], half);
#pragma unroll
            for (int i = 0; i < 2; ++i) {
                ag[i][j] = WMMA_BF16(af[i], bg, ag[i][j]);
                au[i][j] = WMMA_BF16(af[i], bu, au[i][j]);
            }
        }
        __syncthreads();
    }

#pragma unroll
    for (int i = 0; i < 2; ++i) {
#pragma unroll
        for (int j = 0; j < 2; ++j) {
            int n = n0 + wn * 32 + j * 16 + hl;
#pragma unroll
            for (int r = 0; r < 8; ++r) {
                int m = m0 + wm * 32 + i * 16 + half * 8 + r;
                float g = ag[i][j][r], u = au[i][j][r];
                float t = fast_tanh(0.7978845608028654f * (g + 0.044715f * g * g * g));
                float gel = 0.5f * g * (1.f + t);
                outP[(size_t)m * N + n] = (__bf16)(gel * u);
            }
        }
    }
}

// ---------------------------------------------------------------------------
// Flash attention (bf16 in/out, fp32 accumulate). H=2048, HD=128, 8 waves,
// Q-block 128 (16 rows/wave, Q frags in registers), KV-block 64.
// ---------------------------------------------------------------------------
template <bool CAUSAL>
__global__ __launch_bounds__(256) void attn_k(
    const __bf16* __restrict__ Q,   // [B*SQ, 2048]
    const __bf16* __restrict__ Km,  // [B*SKV, 2048]
    const __bf16* __restrict__ Vm,  // [B*SKV, 2048]
    __bf16* __restrict__ O,         // [B*SQ, 2048]
    int SQ, int SKV)
{
    constexpr int HH = 2048, HD = 128;
    __shared__ __bf16 Ks[64 * 128];      // [key][hd]
    __shared__ __bf16 Vs[128 * 64];      // [hd][key]  (transposed)
    __shared__ __bf16 Ps[8 * 16 * 64];   // per-wave [q][key]
    const int tid = threadIdx.x;
    const int wid = tid >> 5, lane = tid & 31;
    const int half = lane >> 4, hl = lane & 15;
    const int qb = blockIdx.x * 128, h = blockIdx.y, b = blockIdx.z;

    // Q fragments in registers (row m = hl per lane, wave owns 16 q rows)
    v16bf aq[4];
    {
        const __bf16* qrow = Q + (size_t)(b * SQ + qb + wid * 16 + hl) * HH + h * HD;
#pragma unroll
        for (int ks = 0; ks < 4; ++ks)
            aq[ks] = load_frag_a(qrow + ks * 32, half);
    }

    v8f co[8] = {};
    float mi[8], li[8];
#pragma unroll
    for (int r = 0; r < 8; ++r) { mi[r] = -1e30f; li[r] = 0.f; }
    const float scale = 0.08838834764831845f;   // 1/sqrt(128)

    const int kend = CAUSAL ? (qb + 128) : SKV;
    for (int kb = 0; kb < kend; kb += 64) {
        // stage K [key][hd] via async global->LDS
#pragma unroll
        for (int it = 0; it < 4; ++it) {
            int id = it * 256 + tid;
            int kk = id >> 4, c = id & 15;
            async_cp16(&Ks[kk * 128 + c * 8],
                       Km + (size_t)(b * SKV + kb + kk) * HH + h * HD + c * 8);
        }
        // stage V transposed -> [hd][key] (through VGPRs)
#pragma unroll
        for (int it = 0; it < 4; ++it) {
            int id = it * 256 + tid;
            int kk = id >> 4, c = id & 15;
            const __bf16* src = Vm + (size_t)(b * SKV + kb + kk) * HH + h * HD + c * 8;
#pragma unroll
            for (int j = 0; j < 8; ++j)
                Vs[(c * 8 + j) * 64 + kk] = src[j];
        }
        async_wait_all();
        __syncthreads();

        // S = Q K^T  (16q x 64key per wave)
        v8f cs[4] = {};
#pragma unroll
        for (int ks = 0; ks < 4; ++ks) {
#pragma unroll
            for (int j = 0; j < 4; ++j) {
                v16bf bk = load_frag_b(&Ks[(j * 16 + hl) * 128 + ks * 32], half);
                cs[j] = WMMA_BF16(aq[ks], bk, cs[j]);
            }
        }
        // scale + mask
#pragma unroll
        for (int j = 0; j < 4; ++j) {
#pragma unroll
            for (int r = 0; r < 8; ++r) {
                float s = cs[j][r] * scale;
                if (CAUSAL) {
                    int key = kb + j * 16 + hl;
                    int q = qb + wid * 16 + half * 8 + r;
                    if (key > q) s -= 1e4f;
                }
                cs[j][r] = s;
            }
        }
        // online softmax
#pragma unroll
        for (int r = 0; r < 8; ++r) {
            float mx = cs[0][r];
#pragma unroll
            for (int j = 1; j < 4; ++j) mx = fmaxf(mx, cs[j][r]);
#pragma unroll
            for (int d = 1; d < 16; d <<= 1) mx = fmaxf(mx, __shfl_xor(mx, d, 32));
            float mnew = fmaxf(mi[r], mx);
            float sum = 0.f;
#pragma unroll
            for (int j = 0; j < 4; ++j) {
                float p = __expf(cs[j][r] - mnew);
                cs[j][r] = p; sum += p;
            }
#pragma unroll
            for (int d = 1; d < 16; d <<= 1) sum += __shfl_xor(sum, d, 32);
            float al = __expf(mi[r] - mnew);
            li[r] = li[r] * al + sum;
            mi[r] = mnew;
#pragma unroll
            for (int jo = 0; jo < 8; ++jo) co[jo][r] *= al;
        }
        // P (C-layout regs) -> per-wave LDS [q][key] for A-layout reload
        __bf16* pw = &Ps[wid * 16 * 64];
#pragma unroll
        for (int j = 0; j < 4; ++j)
#pragma unroll
            for (int r = 0; r < 8; ++r)
                pw[(half * 8 + r) * 64 + j * 16 + hl] = (__bf16)cs[j][r];
        // O += P @ V
#pragma unroll
        for (int ks2 = 0; ks2 < 2; ++ks2) {
            v16bf ap = load_frag_a(&pw[hl * 64 + ks2 * 32], half);
#pragma unroll
            for (int jo = 0; jo < 8; ++jo) {
                v16bf bv = load_frag_b(&Vs[(jo * 16 + hl) * 64 + ks2 * 32], half);
                co[jo] = WMMA_BF16(ap, bv, co[jo]);
            }
        }
        __syncthreads();
    }

    // normalize + store
#pragma unroll
    for (int jo = 0; jo < 8; ++jo) {
#pragma unroll
        for (int r = 0; r < 8; ++r) {
            int row = b * SQ + qb + wid * 16 + half * 8 + r;
            int col = h * HD + jo * 16 + hl;
            O[(size_t)row * HH + col] = (__bf16)(co[jo][r] / li[r]);
        }
    }
}

// ---------------------------------------------------------------------------
// Elementwise kernels
// ---------------------------------------------------------------------------
__global__ __launch_bounds__(256) void ln_k(
    const float* __restrict__ x, const float* __restrict__ w,
    const float* __restrict__ bb, __bf16* __restrict__ out, int Hd)
{
    __shared__ float red[256];
    const int row = blockIdx.x, tid = threadIdx.x;
    const float* xr = x + (size_t)row * Hd;
    float s = 0.f;
    for (int i = tid; i < Hd; i += 256) s += xr[i];
    red[tid] = s; __syncthreads();
    for (int st = 128; st > 0; st >>= 1) {
        if (tid < st) red[tid] += red[tid + st];
        __syncthreads();
    }
    float mean = red[0] / Hd;
    __syncthreads();
    float v = 0.f;
    for (int i = tid; i < Hd; i += 256) { float d = xr[i] - mean; v += d * d; }
    red[tid] = v; __syncthreads();
    for (int st = 128; st > 0; st >>= 1) {
        if (tid < st) red[tid] += red[tid + st];
        __syncthreads();
    }
    float rstd = rsqrtf(red[0] / Hd + 1e-6f);
    for (int i = tid; i < Hd; i += 256)
        out[(size_t)row * Hd + i] = (__bf16)(w[i] * (xr[i] - mean) * rstd + bb[i]);
}

__global__ void rope_k(__bf16* __restrict__ x, int total_pairs)
{
    int idx = blockIdx.x * 256 + threadIdx.x;
    if (idx >= total_pairs) return;
    int j = idx & 63;            // freq index
    int t = idx >> 6;            // m*16 + head
    int head = t & 15, m = t >> 4;
    int s = m & 1023;            // position in sequence (S=1024)
    size_t base = (size_t)m * 2048 + head * 128;
    // inv_freq = 10000^(-j/64) = exp(-j * ln(10000)/64)
    float ang = (float)s * __expf(-(float)j * 0.14391156f);
    float c = __cosf(ang), sn = __sinf(ang);
    float x1 = (float)x[base + j], x2 = (float)x[base + 64 + j];
    x[base + j]      = (__bf16)(x1 * c - x2 * sn);
    x[base + 64 + j] = (__bf16)(x2 * c + x1 * sn);
}

__global__ void cvt_bf16_k(const float* __restrict__ x, __bf16* __restrict__ y, int n)
{
    int i = blockIdx.x * 256 + threadIdx.x;
    if (i < n) y[i] = (__bf16)x[i];
}

__global__ void fuse_k(float* __restrict__ hs, const float* __restrict__ alpha,
                       const float* __restrict__ cross, int n)
{
    int i = blockIdx.x * 256 + threadIdx.x;
    if (i < n) {
        float a = 1.f / (1.f + __expf(-alpha[i]));
        hs[i] = hs[i] + a * cross[i];
    }
}

// ---------------------------------------------------------------------------
// Host orchestration
// ---------------------------------------------------------------------------
extern "C" void kernel_launch(void* const* d_in, const int* in_sizes, int n_in,
                              void* d_out, int out_size, void* d_ws, size_t ws_size,
                              hipStream_t stream)
{
    (void)in_sizes; (void)n_in; (void)out_size; (void)ws_size;
    const float* hidden   = (const float*)d_in[0];
    const float* enc_f    = (const float*)d_in[1];
    /* d_in[2] attention_mask: causal -1e4, applied analytically */
    const float* wq       = (const float*)d_in[3];
    const float* wk       = (const float*)d_in[4];
    const float* wv       = (const float*)d_in[5];
    const float* wo       = (const float*)d_in[6];
    const float* c_attn_w = (const float*)d_in[7];
    const float* c_attn_b = (const float*)d_in[8];
    const float* c_proj_w = (const float*)d_in[9];
    const float* c_proj_b = (const float*)d_in[10];
    const float* vis_w    = (const float*)d_in[11];
    const float* vis_b    = (const float*)d_in[12];
    const float* gate_w   = (const float*)d_in[13];
    const float* up_w     = (const float*)d_in[14];
    const float* down_w   = (const float*)d_in[15];
    const float* ln1w     = (const float*)d_in[16];
    const float* ln1b     = (const float*)d_in[17];
    const float* ln2w     = (const float*)d_in[18];
    const float* ln2b     = (const float*)d_in[19];
    const float* fcw      = (const float*)d_in[20];
    const float* fcb      = (const float*)d_in[21];
    float* out = (float*)d_out;

    const int M = 2048, Hc = 2048, Ic = 8192, Vc = 1024, Menc = 1152;
    const size_t MB = 1024ull * 1024ull;
    char* ws = (char*)d_ws;
    // slot-reused workspace (total 128 MB)
    __bf16* h1       = (__bf16*)(ws + 0);        // ln1 out -> attn_out -> h2
    __bf16* attn_out = h1;
    __bf16* h2       = h1;
    __bf16* qbuf     = (__bf16*)(ws + 8 * MB);   // q -> cq
    __bf16* cq       = qbuf;
    __bf16* kbuf     = (__bf16*)(ws + 16 * MB);  // k -> cross_attn
    __bf16* xattn    = kbuf;
    __bf16* vbuf     = (__bf16*)(ws + 24 * MB);  // v -> hs_bf16
    __bf16* hs_bf    = vbuf;
    float*  hs_f     = (float*)(ws + 32 * MB);   // hs fp32 (fused in-place)
    __bf16* enc_in   = (__bf16*)(ws + 48 * MB);
    __bf16* enc      = (__bf16*)(ws + 51 * MB);
    float*  cross_f  = (float*)(ws + 56 * MB);
    __bf16* cross_bf = (__bf16*)(ws + 72 * MB);
    float*  alpha_f  = (float*)(ws + 80 * MB);
    __bf16* prod     = (__bf16*)(ws + 96 * MB);

    const dim3 g16(16, 16);

    // ---- self-attention ----
    ln_k<<<M, 256, 0, stream>>>(hidden, ln1w, ln1b, h1, Hc);
    gemm_k<true, false, false, false, false, true><<<g16, 256, 0, stream>>>(
        h1, Hc, wq, Hc, nullptr, nullptr, nullptr, nullptr, qbuf, M, Hc, Hc);
    gemm_k<true, false, false, false, false, true><<<g16, 256, 0, stream>>>(
        h1, Hc, wk, Hc, nullptr, nullptr, nullptr, nullptr, kbuf, M, Hc, Hc);
    gemm_k<true, false, false, false, false, true><<<g16, 256, 0, stream>>>(
        h1, Hc, wv, Hc, nullptr, nullptr, nullptr, nullptr, vbuf, M, Hc, Hc);
    rope_k<<<(M * 16 * 64) / 256, 256, 0, stream>>>(qbuf, M * 16 * 64);
    rope_k<<<(M * 16 * 64) / 256, 256, 0, stream>>>(kbuf, M * 16 * 64);
    attn_k<true><<<dim3(8, 16, 2), 256, 0, stream>>>(qbuf, kbuf, vbuf, attn_out, 1024, 1024);
    gemm_k<true, false, false, true, true, true><<<g16, 256, 0, stream>>>(
        attn_out, Hc, wo, Hc, nullptr, hidden, nullptr, hs_f, hs_bf, M, Hc, Hc);

    // ---- visual cross-attention ----
    gemm_k<false, true, false, false, false, true><<<g16, 256, 0, stream>>>(
        hs_bf, Hc, c_attn_w, 3 * Hc, c_attn_b, nullptr, nullptr, nullptr, cq, M, Hc, Hc);
    cvt_bf16_k<<<(Menc * Vc) / 256, 256, 0, stream>>>(enc_f, enc_in, Menc * Vc);
    gemm_k<false, true, false, false, false, true><<<dim3(16, 9), 256, 0, stream>>>(
        enc_in, Vc, vis_w, Hc, vis_b, nullptr, nullptr, nullptr, enc, Menc, Hc, Vc);
    attn_k<false><<<dim3(8, 16, 2), 256, 0, stream>>>(cq, enc, enc, xattn, 1024, 576);
    gemm_k<false, true, false, false, true, true><<<g16, 256, 0, stream>>>(
        xattn, Hc, c_proj_w, Hc, c_proj_b, nullptr, nullptr, cross_f, cross_bf, M, Hc, Hc);

    // ---- gated fusion: alpha = sigmoid([hs,cross] @ W^T + b) ----
    gemm_k<true, false, false, false, true, false><<<g16, 256, 0, stream>>>(
        hs_bf, Hc, fcw, 2 * Hc, nullptr, nullptr, nullptr, alpha_f, nullptr, M, Hc, Hc);
    gemm_k<true, true, true, false, true, false><<<g16, 256, 0, stream>>>(
        cross_bf, Hc, fcw + Hc, 2 * Hc, fcb, nullptr, alpha_f, alpha_f, nullptr, M, Hc, Hc);
    fuse_k<<<(M * Hc) / 256, 256, 0, stream>>>(hs_f, alpha_f, cross_f, M * Hc);

    // ---- MLP ----
    ln_k<<<M, 256, 0, stream>>>(hs_f, ln2w, ln2b, h2, Hc);
    dualgemm_k<<<dim3(128, 16), 256, 0, stream>>>(h2, Hc, gate_w, up_w, Hc, prod, M, Ic, Hc);
    gemm_k<true, false, false, true, true, false><<<g16, 256, 0, stream>>>(
        prod, Ic, down_w, Ic, nullptr, hs_f, nullptr, out, nullptr, M, Hc, Ic);
}
